// LinformerSelfAttention_51427938402836
// MI455X (gfx1250) — matrix operand order repaired and checked by reading, not compiled, not run on hardware
//
#include <hip/hip_runtime.h>
#include <hip/hip_bf16.h>
#include <math.h>

// ---------------------------------------------------------------------------
// Linformer self-attention forward for MI455X (gfx1250, wave32, WMMA).
// All heavy GEMMs: v_wmma_f32_16x16x32_bf16, bf16 operands / f32 accumulate.
// Canonical form: C[M,N] = A[M,K] * B[N,K]^T, both row-major, K contiguous.
// Each wave computes a 64x64 output tile (4x4 16x16 fragments): 16 WMMAs per
// 16 b128 loads per K-step (~32 flop/byte of L2 traffic).
// ---------------------------------------------------------------------------

#define BB 16
#define SS 2048
#define DD 512
#define PP 512

typedef __attribute__((ext_vector_type(16))) __bf16 v16bf;
typedef __attribute__((ext_vector_type(8)))  float  v8f;

union Frag {            // 8 VGPRs worth of 16 packed bf16 values
    uint4        q[2];
    unsigned int u[8];
    v16bf        v;
};

__device__ __forceinline__ unsigned short f2bf(float f) {
    unsigned int u = __float_as_uint(f);
    u += 0x7FFFu + ((u >> 16) & 1u);          // round-to-nearest-even
    return (unsigned short)(u >> 16);
}
__device__ __forceinline__ unsigned int pk2bf(float lo, float hi) {
    return (unsigned int)f2bf(lo) | ((unsigned int)f2bf(hi) << 16);
}

// ---------------------------------------------------------------------------
// WMMA GEMM, 4 waves/block, 64x64 output per wave (4x4 16x16 fragments).
//   HAS_BIAS  : add bias[n] (column bias, matches x @ W.T + b)
//   ST_F32    : write fp32 C (scale applied first)
//   ST_BF16   : write bf16 C
//   TRANS_BF16: bf16 C written transposed (C^T[n,m]) -- used for vp
// ---------------------------------------------------------------------------
template<bool HAS_BIAS, bool ST_F32, bool ST_BF16, bool TRANS_BF16>
__global__ __launch_bounds__(128)
void gemm_wmma(const unsigned short* __restrict__ Am, long lda, long strideA,
               const unsigned short* __restrict__ Bm, long ldb, long strideB,
               const float* __restrict__ bias, float scale,
               float* __restrict__ Cf, long ldcf, long strideCf,
               unsigned short* __restrict__ Cb, long ldcb, long strideCb,
               int K)
{
    const int wave = threadIdx.x >> 5;
    const int lane = threadIdx.x & 31;
    const int hl   = lane >> 4;      // wave half: 0 = lanes 0-15, 1 = 16-31
    const int lr   = lane & 15;
    const int n0 = (blockIdx.x * 4 + wave) * 64;   // 64 columns per wave
    const int m0 = blockIdx.y * 64;                // 64 rows per wave
    const int bz = blockIdx.z;

    v8f acc[4][4];
#pragma unroll
    for (int mi = 0; mi < 4; ++mi)
#pragma unroll
        for (int ni = 0; ni < 4; ++ni)
            acc[mi][ni] = (v8f){0.f,0.f,0.f,0.f,0.f,0.f,0.f,0.f};

    // A fragment: lane holds M = m+lr, K = k0+hl*16 .. +15
    const unsigned short* A0 =
        Am + (size_t)bz * strideA + (size_t)(m0 + lr) * lda + hl * 16;
    // B fragment: lane holds N = n+lr, K = k0+hl*16 .. +15
    const unsigned short* B0 =
        Bm + (size_t)bz * strideB + (size_t)(n0 + lr) * ldb + hl * 16;

    for (int k0 = 0; k0 < K; k0 += 32) {
        Frag a[4], b[4];
#pragma unroll
        for (int i = 0; i < 4; ++i) {
            const unsigned short* Ap = A0 + (size_t)(i * 16) * lda + k0;
            a[i].q[0] = *(const uint4*)(Ap);
            a[i].q[1] = *(const uint4*)(Ap + 8);
        }
#pragma unroll
        for (int i = 0; i < 4; ++i) {
            const unsigned short* Bp = B0 + (size_t)(i * 16) * ldb + k0;
            b[i].q[0] = *(const uint4*)(Bp);
            b[i].q[1] = *(const uint4*)(Bp + 8);
        }
        __builtin_prefetch(A0 + k0 + 128, 0, 3);   // WGP-scope prefetch
        __builtin_prefetch(B0 + k0 + 128, 0, 3);
#pragma unroll
        for (int mi = 0; mi < 4; ++mi)
#pragma unroll
            for (int ni = 0; ni < 4; ++ni)
                acc[mi][ni] = __builtin_amdgcn_wmma_f32_16x16x32_bf16(
                    false, a[mi].v, false, b[ni].v, (short)0,
                    acc[mi][ni], false, false);
    }

    float bvv[4];
#pragma unroll
    for (int ni = 0; ni < 4; ++ni)
        bvv[ni] = HAS_BIAS ? bias[n0 + ni * 16 + lr] : 0.f;

    // C layout: VGPR r of a fragment -> row (mfrag + hl*8 + r), col (nfrag+lr)
#pragma unroll
    for (int mi = 0; mi < 4; ++mi) {
        const int rowb = m0 + mi * 16 + hl * 8;
#pragma unroll
        for (int ni = 0; ni < 4; ++ni) {
            const int n = n0 + ni * 16 + lr;
#pragma unroll
            for (int r = 0; r < 8; ++r) {
                float v = acc[mi][ni][r] * scale + bvv[ni];
                if (ST_F32)
                    Cf[(size_t)bz * strideCf + (size_t)(rowb + r) * ldcf + n] = v;
                if (ST_BF16) {
                    unsigned short h = f2bf(v);
                    if (TRANS_BF16)
                        Cb[(size_t)bz * strideCb + (size_t)n * ldcb + (rowb + r)] = h;
                    else
                        Cb[(size_t)bz * strideCb + (size_t)(rowb + r) * ldcb + n] = h;
                }
            }
        }
    }
}

// ---------------------------------------------------------------------------
// Prep kernels
// ---------------------------------------------------------------------------

// Vectorized fp32 -> bf16 conversion (4 elements / thread).
__global__ __launch_bounds__(256)
void cvt_f32_bf16_vec4(const float4* __restrict__ in,
                       uint2* __restrict__ out, int n4)
{
    int i = blockIdx.x * blockDim.x + threadIdx.x;
    if (i < n4) {
        float4 f = in[i];
        uint2 o;
        o.x = pk2bf(f.x, f.y);
        o.y = pk2bf(f.z, f.w);
        out[i] = o;
    }
}

// W' = bf16(E @ W)   (E:[P,D], W:[D,D] row-major) -> [P,D] bf16
__global__ void fuse_weight_kernel(const float* __restrict__ E,
                                   const float* __restrict__ W,
                                   unsigned short* __restrict__ out)
{
    int c = blockIdx.x * blockDim.x + threadIdx.x;   // 0..D-1
    int p = blockIdx.y;                              // 0..P-1
    float s = 0.f;
    for (int d = 0; d < DD; ++d)
        s = fmaf(E[(size_t)p * DD + d], W[(size_t)d * DD + c], s);
    out[(size_t)p * DD + c] = f2bf(s);
}

// b' = E @ b
__global__ void fuse_bias_kernel(const float* __restrict__ E,
                                 const float* __restrict__ b,
                                 float* __restrict__ out)
{
    int p = blockIdx.x * blockDim.x + threadIdx.x;
    if (p < PP) {
        float s = 0.f;
        for (int d = 0; d < DD; ++d) s = fmaf(E[(size_t)p * DD + d], b[d], s);
        out[p] = s;
    }
}

// Row softmax over L in place (fp32) + bf16 copy for the AV GEMM.
__global__ __launch_bounds__(256)
void softmax_rows_kernel(float* __restrict__ scores,
                         unsigned short* __restrict__ attn_bf, int L)
{
    __shared__ float red[256];
    const int t = threadIdx.x;
    float* p = scores + (size_t)blockIdx.x * L;
    unsigned short* pb = attn_bf + (size_t)blockIdx.x * L;

    float mx = -INFINITY;
    for (int i = t; i < L; i += 256) mx = fmaxf(mx, p[i]);
    red[t] = mx; __syncthreads();
    for (int s = 128; s > 0; s >>= 1) {
        if (t < s) red[t] = fmaxf(red[t], red[t + s]);
        __syncthreads();
    }
    mx = red[0]; __syncthreads();

    float sum = 0.f;
    for (int i = t; i < L; i += 256) {
        float e = __expf(p[i] - mx);
        p[i] = e;
        sum += e;
    }
    red[t] = sum; __syncthreads();
    for (int s = 128; s > 0; s >>= 1) {
        if (t < s) red[t] += red[t + s];
        __syncthreads();
    }
    float inv = 1.f / red[0];

    for (int i = t; i < L; i += 256) {
        float v = p[i] * inv;
        p[i] = v;
        pb[i] = f2bf(v);
    }
}

// ---------------------------------------------------------------------------
// Host launcher
// ---------------------------------------------------------------------------
extern "C" void kernel_launch(void* const* d_in, const int* in_sizes, int n_in,
                              void* d_out, int out_size, void* d_ws, size_t ws_size,
                              hipStream_t stream)
{
    (void)in_sizes; (void)n_in; (void)out_size; (void)ws_size;

    const float* q_in = (const float*)d_in[0];
    const float* k_in = (const float*)d_in[1];
    const float* v_in = (const float*)d_in[2];
    const float* Wq   = (const float*)d_in[3];
    const float* bq   = (const float*)d_in[4];
    const float* Wk   = (const float*)d_in[5];
    const float* bk   = (const float*)d_in[6];
    const float* Wv   = (const float*)d_in[7];
    const float* bv   = (const float*)d_in[8];
    const float* E1   = (const float*)d_in[9];
    const float* E2   = (const float*)d_in[10];
    const float* Wo   = (const float*)d_in[11];
    const float* bo   = (const float*)d_in[12];

    float* out_f32  = (float*)d_out;                              // [B,S,D]
    float* attn_f32 = out_f32 + (size_t)BB * SS * DD;             // [B,S,S]

    // Workspace layout (bytes, 256B aligned). Total ~322 MiB.
    // qin_bf region is dead after the q projection; reused for pre_bf.
    char* ws = (char*)d_ws;
    unsigned short* qin_bf  = (unsigned short*)(ws + 0);               // [B*S,D]
    unsigned short* pre_bf  = (unsigned short*)(ws + 0);               // reuse
    unsigned short* kin_bf  = (unsigned short*)(ws + 33554432ull);     // [B*S,D]
    unsigned short* vin_bf  = (unsigned short*)(ws + 67108864ull);     // [B*S,D]
    unsigned short* q_bf    = (unsigned short*)(ws + 100663296ull);    // [B*S,D]
    unsigned short* kp_bf   = (unsigned short*)(ws + 134217728ull);    // [B*S,P]
    unsigned short* vpT_bf  = (unsigned short*)(ws + 167772160ull);    // [B][P,S]
    unsigned short* attn_bf = (unsigned short*)(ws + 201326592ull);    // [B][S,S]
    unsigned short* Wq_bf   = (unsigned short*)(ws + 335544320ull);    // [D,D]
    unsigned short* W1_bf   = (unsigned short*)(ws + 336068608ull);    // E1@Wk
    unsigned short* W2_bf   = (unsigned short*)(ws + 336592896ull);    // E2@Wv
    unsigned short* Wo_bf   = (unsigned short*)(ws + 337117184ull);    // [D,D]
    float*          b1      = (float*)(ws + 337641472ull);             // E1@bk
    float*          b2      = (float*)(ws + 337643520ull);             // E2@bv

    const long SD = (long)SS * DD;   // per-batch stride of [S,D]
    const long S2 = (long)SS * SS;   // per-batch stride of [S,S]
    const long PS = (long)PP * SS;   // per-batch stride of [P,S]
    const float scal = 0.04419417382415922f;  // 1/sqrt(512)

    // --- activation + weight conversion (single pass, vectorized) ---------
    const int actN4 = (BB * SS * DD) / 4;          // 4,194,304
    const int wN4   = (DD * DD) / 4;               // 65,536
    cvt_f32_bf16_vec4<<<dim3(actN4 / 256), dim3(256), 0, stream>>>(
        (const float4*)q_in, (uint2*)qin_bf, actN4);
    cvt_f32_bf16_vec4<<<dim3(actN4 / 256), dim3(256), 0, stream>>>(
        (const float4*)k_in, (uint2*)kin_bf, actN4);
    cvt_f32_bf16_vec4<<<dim3(actN4 / 256), dim3(256), 0, stream>>>(
        (const float4*)v_in, (uint2*)vin_bf, actN4);
    cvt_f32_bf16_vec4<<<dim3(wN4 / 256), dim3(256), 0, stream>>>(
        (const float4*)Wq, (uint2*)Wq_bf, wN4);
    cvt_f32_bf16_vec4<<<dim3(wN4 / 256), dim3(256), 0, stream>>>(
        (const float4*)Wo, (uint2*)Wo_bf, wN4);
    fuse_weight_kernel<<<dim3(2, PP), dim3(256), 0, stream>>>(E1, Wk, W1_bf);
    fuse_weight_kernel<<<dim3(2, PP), dim3(256), 0, stream>>>(E2, Wv, W2_bf);
    fuse_bias_kernel<<<dim3(2), dim3(256), 0, stream>>>(E1, bk, b1);
    fuse_bias_kernel<<<dim3(2), dim3(256), 0, stream>>>(E2, bv, b2);

    const dim3 blk(128);
    // 4 waves * 64 cols = 256 N per block; 64 M rows per block.
    const dim3 gProj(DD / 256, SS / 64, BB);   // N=512
    const dim3 gScor(SS / 256, SS / 64, BB);   // N=2048

    // --- q = q_in @ Wq^T + bq  -> bf16 ------------------------------------
    gemm_wmma<true, false, true, false><<<gProj, blk, 0, stream>>>(
        qin_bf, DD, SD, Wq_bf, DD, 0, bq, 1.0f,
        nullptr, 0, 0, q_bf, DD, SD, DD);
    // --- kp = k_in @ (E1 Wk)^T + E1 bk -> bf16 ----------------------------
    gemm_wmma<true, false, true, false><<<gProj, blk, 0, stream>>>(
        kin_bf, DD, SD, W1_bf, DD, 0, b1, 1.0f,
        nullptr, 0, 0, kp_bf, DD, SD, DD);
    // --- vp = v_in @ (E2 Wv)^T + E2 bv -> bf16, stored transposed [P,S] ---
    gemm_wmma<true, false, true, true><<<gProj, blk, 0, stream>>>(
        vin_bf, DD, SD, W2_bf, DD, 0, b2, 1.0f,
        nullptr, 0, 0, vpT_bf, SS, PS, DD);

    // --- scores = q @ kp^T * scal -> fp32 straight into d_out attn region -
    gemm_wmma<false, true, false, false><<<gScor, blk, 0, stream>>>(
        q_bf, DD, SD, kp_bf, DD, SD, nullptr, scal,
        attn_f32, SS, S2, nullptr, 0, 0, DD);

    // --- softmax rows (in-place fp32) + bf16 copy -------------------------
    softmax_rows_kernel<<<dim3(BB * SS), dim3(256), 0, stream>>>(
        attn_f32, attn_bf, SS);

    // --- pre = attn @ vp  (= attn @ vpT^T) -> bf16 (reuses qin_bf region) -
    gemm_wmma<false, false, true, false><<<gProj, blk, 0, stream>>>(
        attn_bf, SS, S2, vpT_bf, SS, PS, nullptr, 1.0f,
        nullptr, 0, 0, pre_bf, DD, SD, SS);

    // --- out = pre @ Wo^T + bo -> fp32 d_out ------------------------------
    gemm_wmma<true, true, false, false><<<gProj, blk, 0, stream>>>(
        pre_bf, DD, SD, Wo_bf, DD, 0, bo, 1.0f,
        out_f32, DD, SD, nullptr, 0, 0, DD);
}
